// Model_19971597927196
// MI455X (gfx1250) — matrix operand order repaired and checked by reading, compiled, not run on hardware
//
#include <hip/hip_runtime.h>
#include <hip/hip_bf16.h>

typedef _Float16 half_t;
typedef __attribute__((ext_vector_type(16))) _Float16 v16h;
typedef __attribute__((ext_vector_type(8)))  float    v8f;

#define D_DIM 64
#define DE_DIM 16
#define FW 96            // feature width: 64 (y) + 16 (eigs) + 16 (zero pad) halves
#define FB 192           // feature row bytes (96 * 2)

// ---------------------------------------------------------------- concat/init
__global__ __launch_bounds__(256)
void k_concat(const float* __restrict__ u, const float* __restrict__ it,
              float* __restrict__ xcur, float* __restrict__ out,
              long nuE, long total) {
    long i = (long)blockIdx.x * 256 + threadIdx.x;
    if (i < total) {
        float v = (i < nuE) ? u[i] : it[i - nuE];
        xcur[i] = v;
        out[i]  = v;          // start accumulating mean(embs): out = x0
    }
}

// -------------------------------------------------- layernorm + f16 features
// One wave (32 lanes) per node row. Produces:
//   y  : fp32 normalized row (used by the SpMM scatter)
//   Fa : f16 [0.125*y , exp(lam)*eigs , 0]  (A-side, scales folded in)
//   Fb : f16 [      y ,          eigs , 0]  (B-side)
__global__ __launch_bounds__(256)
void k_ln_feat(const float* __restrict__ x, const float* __restrict__ eigs,
               const float* __restrict__ lambda0, int layer,
               float* __restrict__ y, half_t* __restrict__ Fa,
               half_t* __restrict__ Fb, int N) {
    const int lane = threadIdx.x & 31;
    const int row  = blockIdx.x * 8 + (threadIdx.x >> 5);
    if (row >= N) return;
    const float* xr = x + (size_t)row * D_DIM;
    float a = xr[lane], b = xr[lane + 32];
    float s = a + b;
#pragma unroll
    for (int o = 16; o; o >>= 1) s += __shfl_xor(s, o, 32);
    float mu = s * (1.0f / 64.0f);
    float da = a - mu, db = b - mu;
    float v = da * da + db * db;
#pragma unroll
    for (int o = 16; o; o >>= 1) v += __shfl_xor(v, o, 32);
    float rs = rsqrtf(v * (1.0f / 64.0f) + 1e-5f);
    float ya = da * rs, yb = db * rs;

    float* yr = y + (size_t)row * D_DIM;
    yr[lane] = ya; yr[lane + 32] = yb;

    half_t* fa = Fa + (size_t)row * FW;
    half_t* fb = Fb + (size_t)row * FW;
    fa[lane]      = (half_t)(0.125f * ya);   // fold 1/sqrt(D)=1/8 into A side
    fa[lane + 32] = (half_t)(0.125f * yb);
    fb[lane]      = (half_t)ya;
    fb[lane + 32] = (half_t)yb;

    float elam = __expf(lambda0[layer]);
    if (lane < 16) {                          // cols 64..79 : eig features
        float ev = eigs[(size_t)row * DE_DIM + lane];
        fa[64 + lane] = (half_t)(elam * ev);
        fb[64 + lane] = (half_t)ev;
    } else {                                  // cols 80..95 : zero pad
        fa[64 + lane] = (half_t)0.0f;
        fb[64 + lane] = (half_t)0.0f;
    }
}

// ------------------------------------------------------- WMMA edge scoring
// One wave per group of 16 edges. A row m = Fa[r[m]], B col n = Fb[c[n]].
// D = A x B^T via 3x v_wmma_f32_16x16x32_f16 over K=96; diag(D)[m] =
//   0.125*dot(y[r],y[c]) + exp(lam)*dot(eigs[r],eigs[c])  == fused score.
__global__ __launch_bounds__(256)
void k_score(const half_t* __restrict__ Fa, const half_t* __restrict__ Fb,
             const int* __restrict__ idx_r, const int* __restrict__ idx_c,
             const int* __restrict__ ptype, const float* __restrict__ pw,
             float* __restrict__ ew, float* __restrict__ S0,
             float* __restrict__ S1, long E) {
    const int  lane = threadIdx.x & 31;
    const long g    = ((long)blockIdx.x * 8 + (threadIdx.x >> 5)) * 16;
    if (g >= E) return;                       // wave-uniform: EXEC stays full
    const int  m  = lane & 15;
    long e = g + m; if (e >= E) e = E - 1;    // clamp tail for gathers
    const int  ar = idx_r[e];
    const int  bc = idx_c[e];
    const bool lo = lane < 16;

    const char* arow = (const char*)Fa + (size_t)ar * FB;
    const char* brow = (const char*)Fb + (size_t)bc * FB;

    v8f acc = {};
#pragma unroll
    for (int k = 0; k < 3; ++k) {
        union { uint4 u[2]; v16h h; } A, B;
        // A 16x32 f16 layout: lanes<16 hold K {0..7,16..23}, lanes>=16 {8..15,24..31}
        const char* ab = arow + k * 64;
        A.u[0] = *(const uint4*)(ab + (lo ? 0 : 16));
        A.u[1] = *(const uint4*)(ab + (lo ? 32 : 48));
        // B 32x16 f16: lanes<16 hold col n K {0..7,16..23}, lanes>=16 {8..15,24..31}
        const char* bb = brow + k * 64 + (lo ? 0 : 16);
        B.u[0] = *(const uint4*)(bb);
        B.u[1] = *(const uint4*)(bb + 32);
        acc = __builtin_amdgcn_wmma_f32_16x16x32_f16(
                  false, A.h, false, B.h, (short)0, acc, false, false);
    }

    // Diagonal extraction, one shuffle total.
    // D(m,m): m<8 -> VGPR m @ lane m;  m>=8 -> VGPR m-8 @ lane m+16
    // (lanes 16..31 hold N=lane-16, M=j+8).  First every lane selects
    // acc[lane&7]; at source lane s<8 that is acc[s]=diag comp for edge s,
    // at source lane s in 24..31, lane&7 = s-24 = m-8 = the right component.
    const int l7 = lane & 7;
    float vsel = acc[0];
    vsel = (l7 == 1) ? acc[1] : vsel;
    vsel = (l7 == 2) ? acc[2] : vsel;
    vsel = (l7 == 3) ? acc[3] : vsel;
    vsel = (l7 == 4) ? acc[4] : vsel;
    vsel = (l7 == 5) ? acc[5] : vsel;
    vsel = (l7 == 6) ? acc[6] : vsel;
    vsel = (l7 == 7) ? acc[7] : vsel;
    const int src = (lane < 8) ? lane : (lane + 16);   // lanes 0..15 relevant
    float sc = __shfl(vsel, src, 32);                  // lane m gets diag(m)

    if (lo) {
        long ee = g + lane;
        if (ee < E) {
            // torch clamp(exp(v),-5,5) == min(exp(v),5)
            float e0 = fminf(__expf(sc), 5.0f);
            float e1 = fminf(__expf(pw[ptype[ee]]), 5.0f);
            ew[2 * ee]     = e0;
            ew[2 * ee + 1] = e1;
            atomicAdd(&S0[ar], e0);
            atomicAdd(&S1[ar], e1);
        }
    }
}

// --------------------------------------------------------------- SpMM scatter
// One wave per edge: xnext[r] += 0.5*(e0/S0[r] + e1/S1[r]) * y[c]
__global__ __launch_bounds__(256)
void k_scatter(const float* __restrict__ y, const int* __restrict__ idx_r,
               const int* __restrict__ idx_c, const float* __restrict__ ew,
               const float* __restrict__ S0, const float* __restrict__ S1,
               float* __restrict__ xn, long E) {
    const int  lane = threadIdx.x & 31;
    const long e    = (long)blockIdx.x * 8 + (threadIdx.x >> 5);
    if (e >= E) return;
    int r = idx_r[e], c = idx_c[e];
    float e0 = ew[2 * e], e1 = ew[2 * e + 1];
    float s0 = S0[r], s1 = S1[r];
    s0 = (s0 == 0.0f) ? 1.0f : s0;
    s1 = (s1 == 0.0f) ? 1.0f : s1;
    float s = 0.5f * (e0 / s0 + e1 / s1);
    const float* yc = y  + (size_t)c * D_DIM;
    float*       xr = xn + (size_t)r * D_DIM;
    atomicAdd(&xr[lane],      s * yc[lane]);
    atomicAdd(&xr[lane + 32], s * yc[lane + 32]);
}

__global__ __launch_bounds__(256)
void k_add(float* __restrict__ out, const float* __restrict__ xn, long total) {
    long i = (long)blockIdx.x * 256 + threadIdx.x;
    if (i < total) out[i] += xn[i];
}

__global__ __launch_bounds__(256)
void k_scale(float* __restrict__ out, float f, long total) {
    long i = (long)blockIdx.x * 256 + threadIdx.x;
    if (i < total) out[i] *= f;
}

// ------------------------------------------------------------------- launcher
extern "C" void kernel_launch(void* const* d_in, const int* in_sizes, int n_in,
                              void* d_out, int out_size, void* d_ws, size_t ws_size,
                              hipStream_t stream) {
    (void)n_in; (void)out_size; (void)ws_size;
    const float* user_emb = (const float*)d_in[0];
    const float* item_emb = (const float*)d_in[1];
    const float* eigs     = (const float*)d_in[2];
    const float* lambda0  = (const float*)d_in[3];
    const float* path_w   = (const float*)d_in[4];
    const int*   indices  = (const int*)d_in[5];
    const int*   ptype    = (const int*)d_in[6];

    const long nuE = in_sizes[0];                 // NU*64
    const long N   = (in_sizes[0] + in_sizes[1]) / D_DIM;
    const long E   = in_sizes[5] / 4;             // L*2*E, L=2
    const int  NP  = in_sizes[4] / 2;
    const long total = N * D_DIM;

    // workspace carve-up (256B aligned)
    char* base = (char*)d_ws;
    size_t off = 0;
    auto carve = [&](size_t bytes) { char* p = base + off;
                                     off = (off + bytes + 255) & ~(size_t)255;
                                     return p; };
    float*  y    = (float*)carve(total * 4);
    half_t* Fa   = (half_t*)carve((size_t)N * FW * 2);
    half_t* Fb   = (half_t*)carve((size_t)N * FW * 2);
    float*  xA   = (float*)carve(total * 4);
    float*  xB   = (float*)carve(total * 4);
    float*  S0   = (float*)carve(N * 4);
    float*  S1   = (float*)carve(N * 4);
    float*  ew   = (float*)carve((size_t)E * 2 * 4);
    float*  out  = (float*)d_out;

    float* xc = xA;
    float* xn = xB;

    k_concat<<<(int)((total + 255) / 256), 256, 0, stream>>>(
        user_emb, item_emb, xc, out, nuE, total);

    const int lnBlocks  = (int)((N + 7) / 8);
    const int grpBlocks = (int)(((E + 15) / 16 + 7) / 8);
    const int edBlocks  = (int)((E + 7) / 8);
    const int elBlocks  = (int)((total + 255) / 256);

    for (int l = 0; l < 2; ++l) {
        hipMemsetAsync(S0, 0, N * 4, stream);
        hipMemsetAsync(S1, 0, N * 4, stream);
        hipMemsetAsync(xn, 0, total * 4, stream);

        k_ln_feat<<<lnBlocks, 256, 0, stream>>>(xc, eigs, lambda0, l,
                                                y, Fa, Fb, (int)N);

        const int* ir = indices + (size_t)l * 2 * E;
        const int* ic = ir + E;
        const int* pt = ptype + (size_t)l * E;
        const float* pw = path_w + (size_t)l * NP;

        k_score<<<grpBlocks, 256, 0, stream>>>(Fa, Fb, ir, ic, pt, pw,
                                               ew, S0, S1, E);
        k_scatter<<<edBlocks, 256, 0, stream>>>(y, ir, ic, ew, S0, S1, xn, E);
        k_add<<<elBlocks, 256, 0, stream>>>(out, xn, total);

        float* t = xc; xc = xn; xn = t;           // ping-pong
    }

    k_scale<<<elBlocks, 256, 0, stream>>>(out, 1.0f / 3.0f, total);
}
